// ScaledDotProductWithMapAttention_30262339567908
// MI455X (gfx1250) — matrix operand order repaired and checked by reading, not compiled, not run on hardware
//
#include <hip/hip_runtime.h>
#include <hip/hip_bf16.h>
#include <cmath>

typedef __attribute__((ext_vector_type(16))) __bf16        v16bf;
typedef __attribute__((ext_vector_type(8)))  float         v8f;
typedef __attribute__((ext_vector_type(4)))  unsigned int  u32x4;

#define H_    12
#define DIN   768
#define SEQ   1024
#define DHEAD 64
#define NB    4

// ---- helpers --------------------------------------------------------------

__device__ __forceinline__ unsigned short f2bf(float f) {
  unsigned int u = __float_as_uint(f);
  u += 0x7FFFu + ((u >> 16) & 1u);           // round-to-nearest-even
  return (unsigned short)(u >> 16);
}

// Build a v16bf WMMA operand from two 16-byte chunks (raw bf16 bit patterns).
__device__ __forceinline__ v16bf ld16bf(const unsigned short* p0,
                                        const unsigned short* p1) {
  union { u32x4 u[2]; v16bf v; } t;
  t.u[0] = *(const u32x4*)p0;
  t.u[1] = *(const u32x4*)p1;
  return t.v;
}

__device__ __forceinline__ v8f wmma_bf16(v16bf a, v16bf b, v8f c) {
  return __builtin_amdgcn_wmma_f32_16x16x32_bf16(
      /*neg_a=*/false, a, /*neg_b=*/false, b,
      /*c_mod=*/(short)0, c, /*reuse_a=*/false, /*reuse_b=*/false);
}

// ---- prep kernels ---------------------------------------------------------

__global__ void cvt_bf16_kernel(const float* __restrict__ in,
                                unsigned short* __restrict__ out, int n) {
  for (int i = blockIdx.x * blockDim.x + threadIdx.x; i < n;
       i += gridDim.x * blockDim.x)
    out[i] = f2bf(in[i]);
}

// WT[n*768 + k] = bf16(W[k*768 + n])  (N x K layout for the WMMA B operand)
__global__ void wT_bf16_kernel(const float* __restrict__ W,
                               unsigned short* __restrict__ WT) {
  int i = blockIdx.x * blockDim.x + threadIdx.x;
  if (i < DIN * DIN) {
    int n = i / DIN, k = i - n * DIN;
    WT[i] = f2bf(W[(size_t)k * DIN + n]);
  }
}

// ---- WMMA GEMM: Y = X(rowsx768,bf16) * W(768x768) + bias -----------------
// mode 0: out bf16 laid out (b,h,n,64)       -- for Q and K
// mode 1: out bf16 laid out (b,h,64,n)       -- V transposed, PV B-operand
// mode 2: out f32  laid out (b*n, 768)       -- final output
__global__ __launch_bounds__(32)
void gemm16_kernel(const unsigned short* __restrict__ X,
                   const unsigned short* __restrict__ WT,
                   const float* __restrict__ bias,
                   void* __restrict__ out, int mode) {
  const int lane = threadIdx.x;
  const int half = lane >> 4, l15 = lane & 15;
  const int rowbase = blockIdx.y * 16, colbase = blockIdx.x * 16;

  const unsigned short* xrow = X  + (size_t)(rowbase + l15) * DIN;
  const unsigned short* wrow = WT + (size_t)(colbase + l15) * DIN;

  v8f c = {};
  for (int k0 = 0; k0 < DIN; k0 += 32) {
    // A: lane=M, low half K{0-7,16-23}, high half K{8-15,24-31}
    v16bf a = ld16bf(xrow + k0 + half * 8, xrow + k0 + half * 8 + 16);
    // B: lane=N, low half K{0-7,16-23}, high half K{8-15,24-31}
    v16bf b = ld16bf(wrow + k0 + half * 8, wrow + k0 + half * 8 + 16);
    c = wmma_bf16(a, b, c);
  }

  const int n = colbase + l15;
  const float bvv = bias[n];
  for (int r = 0; r < 8; ++r) {
    float y = c[r] + bvv;                       // C: M = r + 8*half, N = l15
    int m = rowbase + r + half * 8;
    if (mode == 2) {
      ((float*)out)[(size_t)m * DIN + n] = y;
    } else {
      int bb = m >> 10, mm = m & 1023;
      int h = n >> 6, d = n & 63;
      size_t idx = (mode == 0)
        ? ((((size_t)bb * H_ + h) * SEQ + mm) * DHEAD + d)
        : ((((size_t)bb * H_ + h) * DHEAD + d) * SEQ + mm);
      ((unsigned short*)out)[idx] = f2bf(y);
    }
  }
}

// ---- fused flash attention (per b,h,16-row q tile; one wave) --------------

__global__ __launch_bounds__(32)
void attn_kernel(const unsigned short* __restrict__ Q,
                 const unsigned short* __restrict__ K,
                 const unsigned short* __restrict__ Vt,
                 const float* __restrict__ loss_map,
                 const unsigned char* __restrict__ mask,
                 const float* __restrict__ aw,
                 unsigned short* __restrict__ attn_out) {
  const int lane = threadIdx.x, half = lane >> 4, l15 = lane & 15;
  const int qbase = blockIdx.x * 16;
  const int h = blockIdx.y, bb = blockIdx.z;
  const size_t bh = (size_t)bb * H_ + h;
  const int db = half * 8;

  // Q tile (16x64) as two A operands, kept in registers for the whole loop
  const unsigned short* qrow = Q + (bh * SEQ + qbase + l15) * DHEAD;
  v16bf qa0 = ld16bf(qrow + half * 16,      qrow + half * 16 + 8);   // d 0..31
  v16bf qa1 = ld16bf(qrow + half * 16 + 32, qrow + half * 16 + 40);   // d 32..63

  float mrow[8], lrow[8], sc[8];
  v8f acc[4] = {};
  for (int r = 0; r < 8; ++r) { mrow[r] = -INFINITY; lrow[r] = 0.f; }

  __shared__ __align__(64) unsigned short pt[16 * 32];  // P tile 16x32 bf16

  for (int kc = 0; kc < SEQ; kc += 32) {
    // ---- S = Q * K^T for two 16-col k subtiles ----
    v8f s0 = {}, s1 = {};
    {
      const unsigned short* kr0 = K + (bh * SEQ + kc + l15) * DHEAD;
      const unsigned short* kr1 = K + (bh * SEQ + kc + 16 + l15) * DHEAD;
      const int ds0 = half * 16;
      v16bf b00 = ld16bf(kr0 + ds0,      kr0 + ds0 + 8);
      v16bf b01 = ld16bf(kr0 + ds0 + 32, kr0 + ds0 + 40);
      s0 = wmma_bf16(qa0, b00, s0);
      s0 = wmma_bf16(qa1, b01, s0);
      v16bf b10 = ld16bf(kr1 + ds0,      kr1 + ds0 + 8);
      v16bf b11 = ld16bf(kr1 + ds0 + 32, kr1 + ds0 + 40);
      s1 = wmma_bf16(qa0, b10, s1);
      s1 = wmma_bf16(qa1, b11, s1);
    }

    const int kg0 = kc + l15, kg1 = kc + 16 + l15;
    const float lm0 = loss_map[((size_t)bb * SEQ + kg0) * H_ + h];
    const float lm1 = loss_map[((size_t)bb * SEQ + kg1) * H_ + h];

    // ---- scale / weights / mask / bias + online softmax ----
    for (int r = 0; r < 8; ++r) {
      int qg = qbase + r + half * 8;
      size_t ro = (bh * SEQ + qg) * SEQ;
      float v0 = s0[r] * 0.125f * aw[ro + kg0];
      float v1 = s1[r] * 0.125f * aw[ro + kg1];
      if (mask[ro + kg0]) v0 = -INFINITY;
      if (mask[ro + kg1]) v1 = -INFINITY;
      v0 += lm0; v1 += lm1;

      float mx = fmaxf(v0, v1);
      for (int o = 1; o < 16; o <<= 1) mx = fmaxf(mx, __shfl_xor(mx, o, 32));
      float mnew = fmaxf(mrow[r], mx);
      float scale = (mrow[r] == mnew) ? 1.f : __expf(mrow[r] - mnew);
      float p0 = (v0 == -INFINITY) ? 0.f : __expf(v0 - mnew);
      float p1 = (v1 == -INFINITY) ? 0.f : __expf(v1 - mnew);
      float sum = p0 + p1;
      for (int o = 1; o < 16; o <<= 1) sum += __shfl_xor(sum, o, 32);
      lrow[r] = lrow[r] * scale + sum;
      mrow[r] = mnew;
      sc[r] = scale;

      int m = r + half * 8;                    // redistribute via LDS
      pt[m * 32 + l15]      = f2bf(p0);
      pt[m * 32 + 16 + l15] = f2bf(p1);
    }
    __syncthreads();

    for (int dt = 0; dt < 4; ++dt)
      for (int r = 0; r < 8; ++r)
        acc[dt][r] *= sc[r];

    // P re-read in A-operand layout, then O += P * V (K=32)
    v16bf pa = ld16bf(pt + l15 * 32 + half * 16, pt + l15 * 32 + half * 16 + 8);
    for (int dt = 0; dt < 4; ++dt) {
      const unsigned short* vr =
          Vt + (bh * DHEAD + dt * 16 + l15) * SEQ + kc + half * 16;
      v16bf vb = ld16bf(vr, vr + 8);
      acc[dt] = wmma_bf16(pa, vb, acc[dt]);
    }
    __syncthreads();
  }

  // ---- normalize and emit (b, nq, H*64) bf16 for the output projection ----
  for (int dt = 0; dt < 4; ++dt)
    for (int r = 0; r < 8; ++r) {
      int qg = qbase + r + half * 8;
      float o = acc[dt][r] / lrow[r];
      attn_out[((size_t)bb * SEQ + qg) * DIN + h * DHEAD + dt * 16 + l15] =
          f2bf(o);
    }
}

// ---- host launcher --------------------------------------------------------

extern "C" void kernel_launch(void* const* d_in, const int* in_sizes, int n_in,
                              void* d_out, int out_size, void* d_ws,
                              size_t ws_size, hipStream_t stream) {
  (void)in_sizes; (void)n_in; (void)out_size; (void)ws_size;
  const float* queries  = (const float*)d_in[0];
  const float* keys     = (const float*)d_in[1];
  const float* values   = (const float*)d_in[2];
  const float* loss_map = (const float*)d_in[3];
  const unsigned char* amask = (const unsigned char*)d_in[4];
  const float* aw       = (const float*)d_in[5];
  const float* Wq = (const float*)d_in[6];   const float* bq = (const float*)d_in[7];
  const float* Wk = (const float*)d_in[8];   const float* bk = (const float*)d_in[9];
  const float* Wv = (const float*)d_in[10];  const float* bv = (const float*)d_in[11];
  const float* Wo = (const float*)d_in[12];  const float* bo = (const float*)d_in[13];

  const size_t nX = (size_t)NB * SEQ * DIN;   // 3,145,728
  const size_t nW = (size_t)DIN * DIN;        //   589,824

  unsigned short* p   = (unsigned short*)d_ws;
  unsigned short* Xq  = p; p += nX;
  unsigned short* Xk  = p; p += nX;
  unsigned short* Xv  = p; p += nX;
  unsigned short* WTq = p; p += nW;
  unsigned short* WTk = p; p += nW;
  unsigned short* WTv = p; p += nW;
  unsigned short* WTo = p; p += nW;
  unsigned short* Qp  = p; p += nX;
  unsigned short* Kp  = p; p += nX;
  unsigned short* Vt  = p; p += nX;
  unsigned short* Ao  = p; p += nX;

  cvt_bf16_kernel<<<2048, 256, 0, stream>>>(queries, Xq, (int)nX);
  cvt_bf16_kernel<<<2048, 256, 0, stream>>>(keys,    Xk, (int)nX);
  cvt_bf16_kernel<<<2048, 256, 0, stream>>>(values,  Xv, (int)nX);

  const int wblocks = (int)((nW + 255) / 256);
  wT_bf16_kernel<<<wblocks, 256, 0, stream>>>(Wq, WTq);
  wT_bf16_kernel<<<wblocks, 256, 0, stream>>>(Wk, WTk);
  wT_bf16_kernel<<<wblocks, 256, 0, stream>>>(Wv, WTv);
  wT_bf16_kernel<<<wblocks, 256, 0, stream>>>(Wo, WTo);

  dim3 ggrid(DIN / 16, NB * SEQ / 16);   // (48, 256)
  gemm16_kernel<<<ggrid, 32, 0, stream>>>(Xq, WTq, bq, Qp, 0);
  gemm16_kernel<<<ggrid, 32, 0, stream>>>(Xk, WTk, bk, Kp, 0);
  gemm16_kernel<<<ggrid, 32, 0, stream>>>(Xv, WTv, bv, Vt, 1);

  dim3 agrid(SEQ / 16, H_, NB);          // (64, 12, 4)
  attn_kernel<<<agrid, 32, 0, stream>>>(Qp, Kp, Vt, loss_map, amask, aw, Ao);

  gemm16_kernel<<<ggrid, 32, 0, stream>>>(Ao, WTo, bo, d_out, 2);
}
